// LSTMEncoder_71210557767778
// MI455X (gfx1250) — compile-verified, matrix operand order checked by reading, not run on hardware
//
#include <hip/hip_runtime.h>
#include <hip/hip_bf16.h>

// ---------------------------------------------------------------------------
// LSTM encoder, CDNA5 (gfx1250), wave32 + v_wmma_f32_16x16x32_bf16
//
// Fused recurrence:  pre = [x_t | h_prev] @ [[Wi(0:5H) | Ws],[Wi(5H:6H) | 0]]^T
// shapes: A = 512 x 832 (bf16, K padded 812->832), W2 = 3072 x 832 (bf16),
//         pre = 512 x 3072 (f32).  One WMMA GEMM + one gate kernel per step.
//
// GEMM: 64x128 macro-tile / 256-thread block; 8 waves (2x4), each wave does
// 2Mx2N 16x16 tiles (4 WMMA / k-slice, frags reused 2x).  Panels staged into
// double-buffered LDS with GLOBAL_LOAD_ASYNC_TO_LDS_B128 (ASYNCcnt path),
// one split-barrier per k-slice.
// ---------------------------------------------------------------------------

typedef __bf16 bf16;
typedef bf16  v16bf __attribute__((ext_vector_type(16)));
typedef bf16  v8bf  __attribute__((ext_vector_type(8)));
typedef float v8f   __attribute__((ext_vector_type(8)));

#define T_STEPS 128
#define BSZ     512
#define ESZ     300
#define HSZ     512
#define NSZ     3072          // 6*H
#define KPAD    832           // 300 + 512 padded to multiple of 32 (26 k-steps)
#define BM      64
#define BN      128
#define BK      32
#define APAN    (BM * BK)            // 2048 elem = 4 KB
#define BPAN    (BN * BK)            // 4096 elem = 8 KB
#define BUFE    (APAN + BPAN)        // 6144 elem = 12 KB per buffer
#define BUFB    (BUFE * 2)           // bytes per buffer (bf16)

// ---------------- build combined bf16 weight matrix W2 (3072 x 832) --------
__global__ void lstm_init_weights(const float* __restrict__ Wi,
                                  const float* __restrict__ Ws,
                                  bf16* __restrict__ W2) {
    const int total = NSZ * KPAD;
    for (int i = blockIdx.x * blockDim.x + threadIdx.x; i < total;
         i += gridDim.x * blockDim.x) {
        int g = i / KPAD;
        int k = i - g * KPAD;
        float v = 0.0f;
        if (k < ESZ) {
            v = Wi[g * ESZ + k];                    // Wi is (6H, 300) row-major
        } else if (k < ESZ + HSZ && g < 5 * HSZ) {
            v = Ws[g * HSZ + (k - ESZ)];            // Ws is (5H, 512) row-major
        }
        W2[i] = (bf16)v;
    }
}

// ---------------- zero h (d_out), c, and A's h/pad columns ------------------
__global__ void lstm_init_state(float* __restrict__ h,
                                float* __restrict__ c,
                                bf16* __restrict__ A) {
    int i = blockIdx.x * blockDim.x + threadIdx.x;
    if (i < BSZ * HSZ) { h[i] = 0.0f; c[i] = 0.0f; }
    // zero columns 300..831 of A (h portion + K padding) = 512 x 532
    if (i < BSZ * (KPAD - ESZ)) {
        int b = i / (KPAD - ESZ);
        int k = ESZ + (i - b * (KPAD - ESZ));
        A[b * KPAD + k] = (bf16)0.0f;
    }
}

// ---------------- per-step: convert x[t] (512 x 300) into A columns 0..299 --
__global__ void lstm_prep_x(const float* __restrict__ x,
                            bf16* __restrict__ A, int t) {
    int i = blockIdx.x * blockDim.x + threadIdx.x;
    if (i < BSZ * ESZ) {
        int b = i / ESZ;
        int e = i - b * ESZ;
        A[b * KPAD + e] = (bf16)x[(size_t)t * BSZ * ESZ + i];
    }
}

// ---------------- async 16B copy: global -> LDS (ASYNCcnt tracked) ----------
__device__ __forceinline__ void async_cp16(unsigned lds_addr, const bf16* g) {
    asm volatile("global_load_async_to_lds_b128 %0, %1, off"
                 :: "v"(lds_addr), "v"((unsigned long long)(size_t)g)
                 : "memory");
}
__device__ __forceinline__ void wait_async0() {
    asm volatile("s_wait_asynccnt 0x0" ::: "memory");
}

// ---------------- fragment gather per ISA 7.12.2 (wave32) -------------------
// lane l: row (l&15) of the tile, K chunks at (l>>4)*8 and +16 (two 16B loads)
__device__ __forceinline__ v16bf load_frag(const bf16* p) {
    v8bf lo = *(const v8bf*)p;
    v8bf hi = *(const v8bf*)(p + 16);
    v16bf f;
    #pragma unroll
    for (int i = 0; i < 8; ++i) { f[i] = lo[i]; f[i + 8] = hi[i]; }
    return f;
}

// ---------------- WMMA GEMM: pre(512x3072) = A(512x832) * W2(3072x832)^T ----
__global__ void __launch_bounds__(256)
lstm_gemm(const bf16* __restrict__ A, const bf16* __restrict__ W2,
          float* __restrict__ pre) {
    __shared__ __align__(16) bf16 smem[2 * BUFE];   // ping-pong, 24 KB

    const int tid  = threadIdx.x;
    const int wave = tid >> 5;        // 0..7
    const int lane = tid & 31;
    const int half = lane >> 4;       // 0 or 1
    const int r    = lane & 15;

    const int bm = blockIdx.x;        // 0..7   (512/64)
    const int bn = blockIdx.y;        // 0..23  (3072/128)

    const int wm = wave >> 2;         // 0..1 -> pair of M tiles
    const int wn = wave & 3;          // 0..3 -> pair of N tiles

    // staging: each thread async-copies 3 x 16B chunks per k-slice
    //   A panel: chunk tid        -> row tid>>2,        col (tid&3)*8
    //   B panel: chunks tid,tid+256 -> rows tid>>2, 64+(tid>>2), same col
    const int prow = tid >> 2;            // 0..63
    const int pcol = (tid & 3) * 8;       // 0,8,16,24
    const bf16* gA  = A  + (size_t)(bm * BM + prow) * KPAD + pcol;
    const bf16* gB0 = W2 + (size_t)(bn * BN + prow) * KPAD + pcol;
    const bf16* gB1 = W2 + (size_t)(bn * BN + 64 + prow) * KPAD + pcol;

    const unsigned ldsBase = (unsigned)(size_t)(&smem[0]);
    const unsigned pOff    = (unsigned)((prow * BK + pcol) * sizeof(bf16));
    const unsigned aOff    = ldsBase + pOff;                     // A chunk
    const unsigned bOff0   = ldsBase + APAN * 2 + pOff;          // B rows 0..63
    const unsigned bOff1   = bOff0 + 64 * BK * 2;                // B rows 64..127

    v8f acc00 = {}, acc01 = {}, acc10 = {}, acc11 = {};

    // prologue: stage k-slice 0 into buffer 0
    async_cp16(aOff,  gA);
    async_cp16(bOff0, gB0);
    async_cp16(bOff1, gB1);

    int buf = 0;
    for (int k0 = 0; k0 < KPAD; k0 += BK) {
        wait_async0();          // this wave's panel copies are in LDS
        __syncthreads();        // everyone's copies visible; prev reads done

        if (k0 + BK < KPAD) {   // stage next slice into the other buffer
            const unsigned nb = (buf ^ 1) * (unsigned)BUFB;
            async_cp16(aOff  + nb, gA  + k0 + BK);
            async_cp16(bOff0 + nb, gB0 + k0 + BK);
            async_cp16(bOff1 + nb, gB1 + k0 + BK);
        }

        // fragments from current buffer (ds_load_b128 x8)
        const bf16* sbuf = smem + buf * BUFE;
        const bf16* sA   = sbuf;
        const bf16* sB   = sbuf + APAN;
        v16bf a0 = load_frag(sA + (wm * 32 + r)      * BK + half * 8);
        v16bf a1 = load_frag(sA + (wm * 32 + 16 + r) * BK + half * 8);
        v16bf b0 = load_frag(sB + (wn * 32 + r)      * BK + half * 8);
        v16bf b1 = load_frag(sB + (wn * 32 + 16 + r) * BK + half * 8);

        acc00 = __builtin_amdgcn_wmma_f32_16x16x32_bf16(
                    false, a0, false, b0, (short)0, acc00, false, false);
        acc01 = __builtin_amdgcn_wmma_f32_16x16x32_bf16(
                    false, a0, false, b1, (short)0, acc01, false, false);
        acc10 = __builtin_amdgcn_wmma_f32_16x16x32_bf16(
                    false, a1, false, b0, (short)0, acc10, false, false);
        acc11 = __builtin_amdgcn_wmma_f32_16x16x32_bf16(
                    false, a1, false, b1, (short)0, acc11, false, false);
        buf ^= 1;
    }

    // D layout: lane l holds col (l&15); rows (l>>4)*8 + v, VGPR v = 0..7
    const int col0 = bn * BN + wn * 32 + r;
    const int row0 = bm * BM + wm * 32 + half * 8;
    float* cp = pre + (size_t)row0 * NSZ + col0;
    #pragma unroll
    for (int v = 0; v < 8; ++v) {
        cp[(size_t)v * NSZ]             = acc00[v];
        cp[(size_t)v * NSZ + 16]        = acc01[v];
        cp[(size_t)(v + 16) * NSZ]      = acc10[v];
        cp[(size_t)(v + 16) * NSZ + 16] = acc11[v];
    }
}

// ---------------- per-step gates + masked state update ----------------------
__device__ __forceinline__ float sigmoidf_fast(float x) {
    return 1.0f / (1.0f + __expf(-x));
}

__global__ void lstm_gates(const float* __restrict__ pre,
                           const float* __restrict__ bi,
                           const float* __restrict__ bs,
                           const int*   __restrict__ lengths,
                           float* __restrict__ h,    // d_out, (512,512)
                           float* __restrict__ c,
                           bf16*  __restrict__ A,    // writes h columns 300..811
                           int t) {
    int i = blockIdx.x * blockDim.x + threadIdx.x;
    if (i >= BSZ * HSZ) return;
    int b = i >> 9;            // /512
    int j = i & (HSZ - 1);     // %512

    const float* row = pre + (size_t)b * NSZ;
    float zi = row[j]            + bi[j]            + bs[j];
    float zf = row[HSZ + j]      + bi[HSZ + j]      + bs[HSZ + j];
    float zg = row[2 * HSZ + j]  + bi[2 * HSZ + j]  + bs[2 * HSZ + j];
    float zo = row[3 * HSZ + j]  + bi[3 * HSZ + j]  + bs[3 * HSZ + j];
    float zw = row[4 * HSZ + j]  + bi[4 * HSZ + j]  + bs[4 * HSZ + j];
    float zp = row[5 * HSZ + j]  + bi[5 * HSZ + j];   // pi_proj (no ps term)

    float ig = sigmoidf_fast(zi);
    float fg = sigmoidf_fast(zf);
    float gg = tanhf(zg);
    float og = sigmoidf_fast(zo);
    float hw = sigmoidf_fast(zw);

    float c_prev = c[i];
    float h_prev = h[i];
    float c_new  = ig * gg + fg * c_prev;
    float out    = og * tanhf(c_new);
    out = hw * out + (1.0f - hw) * zp;

    bool valid = t < lengths[b];
    float h_n = valid ? out   : h_prev;
    float c_n = valid ? c_new : c_prev;

    h[i] = h_n;
    c[i] = c_n;
    A[(size_t)b * KPAD + ESZ + j] = (bf16)h_n;   // feed next step's GEMM
}

// ---------------------------------------------------------------------------
extern "C" void kernel_launch(void* const* d_in, const int* in_sizes, int n_in,
                              void* d_out, int out_size, void* d_ws, size_t ws_size,
                              hipStream_t stream) {
    const float* x       = (const float*)d_in[0];   // (128, 512, 300)
    const int*   lengths = (const int*)  d_in[1];   // (512,)
    const float* Wi      = (const float*)d_in[2];   // (3072, 300)
    const float* bi      = (const float*)d_in[3];   // (3072,)
    const float* Ws      = (const float*)d_in[4];   // (2560, 512)
    const float* bs      = (const float*)d_in[5];   // (2560,)
    float*       hout    = (float*)d_out;           // (512, 512)

    char* ws = (char*)d_ws;
    const size_t W2_BYTES  = (size_t)NSZ * KPAD * sizeof(bf16);   // 5,111,808
    const size_t A_BYTES   = (size_t)BSZ * KPAD * sizeof(bf16);   //   851,968
    const size_t PRE_BYTES = (size_t)BSZ * NSZ * sizeof(float);   // 6,291,456
    bf16*  W2  = (bf16*)ws;
    bf16*  Amx = (bf16*)(ws + W2_BYTES);
    float* pre = (float*)(ws + W2_BYTES + A_BYTES);
    float* cbf = (float*)(ws + W2_BYTES + A_BYTES + PRE_BYTES);
    (void)in_sizes; (void)n_in; (void)out_size; (void)ws_size;

    // Build fused bf16 weights; zero h, c and A's h/pad columns.
    lstm_init_weights<<<1024, 256, 0, stream>>>(Wi, Ws, W2);
    {
        int n = BSZ * (KPAD - ESZ);            // 272384 > 262144
        lstm_init_state<<<(n + 255) / 256, 256, 0, stream>>>(hout, cbf, Amx);
    }

    const dim3 gemm_grid(BSZ / BM, NSZ / BN);          // 8 x 24 = 192 blocks
    const int  prep_blocks = (BSZ * ESZ + 255) / 256;  // 600
    const int  gate_blocks = (BSZ * HSZ + 255) / 256;  // 1024

    for (int t = 0; t < T_STEPS; ++t) {
        lstm_prep_x<<<prep_blocks, 256, 0, stream>>>(x, Amx, t);
        lstm_gemm  <<<gemm_grid,   256, 0, stream>>>(Amx, W2, pre);
        lstm_gates <<<gate_blocks, 256, 0, stream>>>(pre, bi, bs, lengths,
                                                     hout, cbf, Amx, t);
    }
}
